// BarrierNet_52158082842728
// MI455X (gfx1250) — matrix operand order repaired and checked by reading, not compiled, not run on hardware
//
#include <hip/hip_runtime.h>
#include <stdint.h>
#include <stddef.h>

// ---------------------------------------------------------------------------
// BarrierNet on MI455X (gfx1250, wave32)
//   stage 1: h1 = relu((x*std+mean)@W1 + b1)            -> bf16 workspace
//   stage 2: x21/x22 = relu(h1@W2x + b2x) via v_wmma_f32_16x16x32_bf16
//            with 128x128 block tiles (L2 traffic ~256MB) and a one-deep
//            software pipeline (prefetch next K-step's fragments before
//            issuing current WMMAs), fused bias+relu+head-GEMV epilogue,
//            deterministic per-block head partials (graph-replay safe)
//   stage 3: per-row 9-constraint QP, 300 projected-gradient dual iterations
// Dead branches of the reference (x23/W33/b33, sgn) are eliminated.
// ---------------------------------------------------------------------------

#define FEAT 8
#define H1DIM 1024
#define H2DIM 512
#define MCON 9
#define QP_ITERS 300
#define BM 128          // rows per block tile
#define NCB 8           // column-blocks: 2 branches x 4 blocks of 128 cols

typedef __attribute__((ext_vector_type(16))) __bf16 v16bf;
typedef __attribute__((ext_vector_type(8)))  float  v8f;
typedef __attribute__((ext_vector_type(4)))  unsigned int u32x4;

__device__ __forceinline__ unsigned short f32_to_bf16(float f) {
    unsigned int u = __float_as_uint(f);
    u += 0x7FFFu + ((u >> 16) & 1u);   // round-to-nearest-even
    return (unsigned short)(u >> 16);
}

// --------------------------------------------------------------------------
// Kernel 0: W2 [H1,H2] f32 (row-major i,o)  ->  Wt [H2,H1] bf16 (col-major)
// so B-fragments become two contiguous 16B reads per lane.
// --------------------------------------------------------------------------
__global__ void transpose_to_bf16(const float* __restrict__ w,
                                  unsigned short* __restrict__ wt) {
    int tid = blockIdx.x * blockDim.x + threadIdx.x;   // 0 .. H2*H1-1
    int n = tid >> 10;        // / H1DIM
    int k = tid & (H1DIM - 1);
    wt[tid] = f32_to_bf16(w[k * H2DIM + n]);
}

// --------------------------------------------------------------------------
// Kernel 1: K=8 input layer (memory-bound) -> bf16 h1
// --------------------------------------------------------------------------
__global__ void layer1_kernel(const float* __restrict__ x,
                              const float* __restrict__ mean,
                              const float* __restrict__ stdv,
                              const float* __restrict__ W1,
                              const float* __restrict__ b1,
                              unsigned short* __restrict__ h1,
                              int nrows) {
    int tid = blockIdx.x * blockDim.x + threadIdx.x;
    int b = tid >> 10;
    int j = tid & (H1DIM - 1);
    if (b >= nrows) return;
    float s = b1[j];
#pragma unroll
    for (int k = 0; k < FEAT; ++k) {
        float xv = fmaf(x[b * FEAT + k], stdv[k], mean[k]);
        s = fmaf(xv, W1[k * H1DIM + j], s);
    }
    h1[tid] = f32_to_bf16(fmaxf(s, 0.0f));
}

// --------------------------------------------------------------------------
// Kernel 2: bf16 WMMA GEMM with 128(M) x 128(N) block tiles.
//   grid = (8 col-blocks, B/128 row-blocks); col-blocks 0-3 -> branch W21,
//   4-7 -> branch W22. 16 waves = 4 (M-groups) x 4 (N-groups); each wave
//   owns a 2x2 grid of 16x16 output tiles (4 accumulators).
//   One-deep software pipeline: fragments for K-step k+32 are loaded into a
//   second register set before the WMMAs of step k are issued, so every
//   s_wait_loadcnt has a full K-step of load latency already hidden.
// --------------------------------------------------------------------------
__global__ __launch_bounds__(512)
void gemm2_heads_kernel(const unsigned short* __restrict__ h1,
                        const unsigned short* __restrict__ w21t,
                        const unsigned short* __restrict__ w22t,
                        const float* __restrict__ b21,
                        const float* __restrict__ b22,
                        const float* __restrict__ W31,
                        const float* __restrict__ W32,
                        float* __restrict__ phead) {   // [B][NCB][2]
    __shared__ float lds_heads[BM][2];

    const int lane = threadIdx.x & 31;
    const int wave = threadIdx.x >> 5;
    const int lm = lane & 15;            // row/col within 16-tile
    const int lh = lane >> 4;            // lane half
    const int cb = blockIdx.x;           // 0..7
    const int mbase = blockIdx.y * BM;
    const int branch = cb >> 2;          // 0 -> W21/W31, 1 -> W22/W32
    const int colbase = (cb & 3) * 128;  // within the branch's 512 columns

    const unsigned short* wt  = branch ? w22t : w21t;
    const float*          bsv = branch ? b22  : b21;
    const float*          hw  = branch ? W32  : W31;

    if (threadIdx.x < BM * 2) ((float*)lds_heads)[threadIdx.x] = 0.0f;
    __syncthreads();

    const int mtg = wave >> 2;           // 0..3  (32-row group)
    const int ctg = wave & 3;            // 0..3  (32-col group)

    // A fragments: rows; B fragments: columns (ISA 16-bit layouts).
    const int row0 = mbase + mtg * 32 + lm;
    const int col0 = colbase + ctg * 32 + lm;
    const unsigned short* pa0 = h1 + (size_t)row0 * H1DIM + (lh << 3);
    const unsigned short* pa1 = pa0 + (size_t)16 * H1DIM;
    const unsigned short* pb0 = wt + (size_t)col0 * H1DIM + (lh << 3);
    const unsigned short* pb1 = pb0 + (size_t)16 * H1DIM;

    v8f acc00 = {0.f,0.f,0.f,0.f,0.f,0.f,0.f,0.f};
    v8f acc01 = acc00, acc10 = acc00, acc11 = acc00;

    union Frag { u32x4 q[2]; v16bf v; };

    // A and B chunks at K = kh*8 and kh*8+16.
#define LOAD_STEP(FA0, FA1, FB0, FB1, KB)                                     \
    do {                                                                      \
        FA0.q[0] = *(const u32x4*)(pa0 + (KB));                               \
        FA0.q[1] = *(const u32x4*)(pa0 + (KB) + 16);                          \
        FA1.q[0] = *(const u32x4*)(pa1 + (KB));                               \
        FA1.q[1] = *(const u32x4*)(pa1 + (KB) + 16);                          \
        FB0.q[0] = *(const u32x4*)(pb0 + (KB));                               \
        FB0.q[1] = *(const u32x4*)(pb0 + (KB) + 16);                          \
        FB1.q[0] = *(const u32x4*)(pb1 + (KB));                               \
        FB1.q[1] = *(const u32x4*)(pb1 + (KB) + 16);                          \
    } while (0)

#define WMMA_STEP(FA0, FA1, FB0, FB1)                                         \
    do {                                                                      \
        acc00 = __builtin_amdgcn_wmma_f32_16x16x32_bf16(                      \
            false, FA0.v, false, FB0.v, (short)0, acc00, false, false);       \
        acc01 = __builtin_amdgcn_wmma_f32_16x16x32_bf16(                      \
            false, FA0.v, false, FB1.v, (short)0, acc01, false, false);       \
        acc10 = __builtin_amdgcn_wmma_f32_16x16x32_bf16(                      \
            false, FA1.v, false, FB0.v, (short)0, acc10, false, false);       \
        acc11 = __builtin_amdgcn_wmma_f32_16x16x32_bf16(                      \
            false, FA1.v, false, FB1.v, (short)0, acc11, false, false);       \
    } while (0)

    Frag ca0, ca1, cb0f, cb1f;           // current K-step fragments
    LOAD_STEP(ca0, ca1, cb0f, cb1f, 0);  // prologue

#pragma unroll 2
    for (int kb = 0; kb < H1DIM - 32; kb += 32) {
        Frag na0, na1, nb0, nb1;         // next K-step fragments (prefetch)
        LOAD_STEP(na0, na1, nb0, nb1, kb + 32);
        WMMA_STEP(ca0, ca1, cb0f, cb1f);
        ca0 = na0; ca1 = na1; cb0f = nb0; cb1f = nb1;   // renamed by unroll
    }
    WMMA_STEP(ca0, ca1, cb0f, cb1f);     // epilogue step

#undef LOAD_STEP
#undef WMMA_STEP

    // Fused epilogue: bias+relu, per-lane head GEMV partials, shuffle-reduce
    // across the 16-lane half (N), ds_add_f32 across waves sharing a row.
    const float bias0 = bsv[col0], bias1 = bsv[col0 + 16];
    const float wA0 = hw[col0 * 2 + 0],        wB0 = hw[col0 * 2 + 1];
    const float wA1 = hw[(col0 + 16) * 2 + 0], wB1 = hw[(col0 + 16) * 2 + 1];

#pragma unroll
    for (int mi = 0; mi < 2; ++mi) {
        const v8f& a0 = mi ? acc10 : acc00;
        const v8f& a1 = mi ? acc11 : acc01;
#pragma unroll
        for (int i = 0; i < 8; ++i) {
            // C/D layout: VGPR i, lanes 0-15 -> row i; lanes 16-31 -> row i+8.
            float v0 = fmaxf(a0[i] + bias0, 0.f);
            float v1 = fmaxf(a1[i] + bias1, 0.f);
            float pA = fmaf(v1, wA1, v0 * wA0);
            float pB = fmaf(v1, wB1, v0 * wB0);
#pragma unroll
            for (int m = 1; m < 16; m <<= 1) {
                pA += __shfl_xor(pA, m, 32);
                pB += __shfl_xor(pB, m, 32);
            }
            if (lm == 0) {
                int row = mtg * 32 + mi * 16 + i + (lh << 3);
                atomicAdd(&lds_heads[row][0], pA);   // ds_add_f32
                atomicAdd(&lds_heads[row][1], pB);
            }
        }
    }
    __syncthreads();
    if (threadIdx.x < BM * 2) {
        int row = threadIdx.x >> 1, h = threadIdx.x & 1;
        phead[((size_t)(mbase + row) * NCB + cb) * 2 + h] = lds_heads[row][h];
    }
}

// --------------------------------------------------------------------------
// Kernel 3: sum head partials (fixed order -> deterministic), build the 9
// barrier constraints, run the dual projected-gradient QP fully in registers.
// --------------------------------------------------------------------------
__global__ __launch_bounds__(256)
void qp_kernel(const float* __restrict__ x,
               const float* __restrict__ mean,
               const float* __restrict__ stdv,
               const float* __restrict__ b31,
               const float* __restrict__ b32,
               const float* __restrict__ obstacles,   // [8][3] = x,y,r
               const float* __restrict__ phead,       // [B][NCB][2]
               float* __restrict__ out,               // [B][2]
               int nrows) {
    int b = blockIdx.x * blockDim.x + threadIdx.x;
    if (b >= nrows) return;

    // Sum the 8 column-block partials: cb 0-3 -> x31, cb 4-7 -> x32 pre-act.
    const float* ph = phead + (size_t)b * NCB * 2;
    float s0 = 0.f, s1 = 0.f, s2 = 0.f, s3 = 0.f;
#pragma unroll
    for (int cbk = 0; cbk < 4; ++cbk) {
        s0 += ph[cbk * 2 + 0];
        s1 += ph[cbk * 2 + 1];
        s2 += ph[(cbk + 4) * 2 + 0];
        s3 += ph[(cbk + 4) * 2 + 1];
    }
    const float pu0 = s0 + b31[0];
    const float pu1 = s1 + b31[1];
    const float pp1 = 4.0f / (1.0f + expf(-(s2 + b32[0])));
    const float pp2 = 4.0f / (1.0f + expf(-(s3 + b32[1])));

    float x0[6];
#pragma unroll
    for (int i = 0; i < 6; ++i)
        x0[i] = fmaf(x[b * FEAT + i], stdv[i], mean[i]);
    const float px = x0[0], py = x0[1], th = x0[2], v = x0[3];
    const float st = sinf(th), ct = cosf(th);

    float G1[MCON], G2[MCON], hv[MCON];
#pragma unroll
    for (int m = 0; m < MCON; ++m) {
        float ox, oy, orad;
        if (m < 8) { ox = obstacles[m*3]; oy = obstacles[m*3+1]; orad = obstacles[m*3+2]; }
        else       { ox = x0[4]; oy = x0[5]; orad = 0.5f; }
        const float R   = 0.5f + orad + 0.1f;       // AGENT_RADIUS + r + SAFETY
        const float dx  = px - ox, dy = py - oy;
        const float bar = dx*dx + dy*dy - R*R;
        const float dct = fmaf(dx, ct, dy * st);
        const float bdot = 2.0f * v * dct;
        G1[m] = 2.0f * v * fmaf(dx, st, -dy * ct);  // -LgLfbu1
        G2[m] = -2.0f * dct;                        // -LgLfbu2
        hv[m] = fmaf(pp1 + pp2, bdot, fmaf(pp1 * pp2, bar, 2.0f * v * v));
    }

    float GGt[MCON][MCON], q[MCON];
    float ss = 0.f;
#pragma unroll
    for (int m = 0; m < MCON; ++m) {
#pragma unroll
        for (int n = 0; n < MCON; ++n) {
            float g = fmaf(G1[m], G1[n], G2[m] * G2[n]);
            GGt[m][n] = g;
            ss = fmaf(g, g, ss);
        }
        q[m] = fmaf(G1[m], pu0, fmaf(G2[m], pu1, hv[m]));
    }
    const float alpha = 1.0f / (sqrtf(ss) + 1e-6f);

    float lam[MCON];
#pragma unroll
    for (int m = 0; m < MCON; ++m) lam[m] = 0.f;

    for (int it = 0; it < QP_ITERS; ++it) {
        float gr[MCON];
#pragma unroll
        for (int m = 0; m < MCON; ++m) {
            float g = q[m];
#pragma unroll
            for (int n = 0; n < MCON; ++n) g = fmaf(GGt[m][n], lam[n], g);
            gr[m] = g;
        }
#pragma unroll
        for (int m = 0; m < MCON; ++m)
            lam[m] = fmaxf(fmaf(-alpha, gr[m], lam[m]), 0.f);
    }

    float u0 = -pu0, u1 = -pu1;
#pragma unroll
    for (int m = 0; m < MCON; ++m) {
        u0 = fmaf(-G1[m], lam[m], u0);
        u1 = fmaf(-G2[m], lam[m], u1);
    }
    out[b * 2 + 0] = u0;
    out[b * 2 + 1] = u1;
}

// --------------------------------------------------------------------------
extern "C" void kernel_launch(void* const* d_in, const int* in_sizes, int n_in,
                              void* d_out, int out_size, void* d_ws, size_t ws_size,
                              hipStream_t stream) {
    const float* x    = (const float*)d_in[0];
    const float* mean = (const float*)d_in[1];
    const float* stdv = (const float*)d_in[2];
    const float* W1   = (const float*)d_in[3];
    const float* b1   = (const float*)d_in[4];
    const float* W21  = (const float*)d_in[5];
    const float* b21  = (const float*)d_in[6];
    const float* W22  = (const float*)d_in[7];
    const float* b22  = (const float*)d_in[8];
    // d_in[9]/[10]  = W23/b23  : dead in reference output, skipped
    const float* W31  = (const float*)d_in[11];
    const float* b31  = (const float*)d_in[12];
    const float* W32  = (const float*)d_in[13];
    const float* b32  = (const float*)d_in[14];
    // d_in[15]/[16] = W33/b33  : dead, skipped
    const float* obstacles = (const float*)d_in[17];
    // d_in[18] = sgn : unused by reference

    const int B = in_sizes[0] / FEAT;   // 8192

    // Workspace: h1 bf16 (16MB) | W21t bf16 (1MB) | W22t bf16 (1MB)
    //            | head partials f32 [B][8][2] (512KB)
    char* ws = (char*)d_ws;
    unsigned short* h1   = (unsigned short*)ws;
    unsigned short* w21t = (unsigned short*)(ws + (size_t)B * H1DIM * 2);
    unsigned short* w22t = (unsigned short*)(ws + (size_t)B * H1DIM * 2
                                                + (size_t)H1DIM * H2DIM * 2);
    float* phead = (float*)(ws + (size_t)B * H1DIM * 2
                               + 2 * (size_t)H1DIM * H2DIM * 2);

    transpose_to_bf16<<<(H1DIM * H2DIM + 255) / 256, 256, 0, stream>>>(W21, w21t);
    transpose_to_bf16<<<(H1DIM * H2DIM + 255) / 256, 256, 0, stream>>>(W22, w22t);
    layer1_kernel<<<(B * H1DIM + 255) / 256, 256, 0, stream>>>(
        x, mean, stdv, W1, b1, h1, B);
    dim3 ggrid(NCB, B / BM);
    gemm2_heads_kernel<<<ggrid, 512, 0, stream>>>(
        h1, w21t, w22t, b21, b22, W31, W32, phead);
    qp_kernel<<<(B + 255) / 256, 256, 0, stream>>>(
        x, mean, stdv, b31, b32, obstacles, phead, (float*)d_out, B);
}
